// GroupedQueryAttention_87471303950372
// MI455X (gfx1250) — matrix operand order repaired and checked by reading, not compiled, not run on hardware
//
#include <hip/hip_runtime.h>
#include <hip/hip_bf16.h>
#include <math.h>
#include <cstdint>

typedef __attribute__((ext_vector_type(16))) _Float16 v16h;
typedef __attribute__((ext_vector_type(8)))  _Float16 v8h;
typedef __attribute__((ext_vector_type(4)))  _Float16 v4h;
typedef __attribute__((ext_vector_type(2)))  _Float16 v2h;
typedef __attribute__((ext_vector_type(8)))  float    v8f;
typedef __attribute__((ext_vector_type(4)))  float    v4f;
typedef unsigned int u32;
typedef __attribute__((ext_vector_type(4))) u32 u32x4;
typedef __attribute__((ext_vector_type(4))) int  i32x4;
typedef __attribute__((ext_vector_type(8))) int  i32x8;

// ---------------------------------------------------------------------------
// Tensor Data Mover: 2D tile load Global(f16) -> LDS with row padding.
// D# built per CDNA5 ISA 8.3/8.4. tensor dims == tile dims (tiles always
// in-bounds), data_size=1 (2 bytes), pad after each row to keep the LDS row
// stride 16B-aligned and bank-conflict-friendly.
// Toolchain arity hedge: therock headers ship amd_gfx1250_TDM.h (6-arg form).
// ---------------------------------------------------------------------------
#if defined(__has_include)
#if __has_include(<hip/amd_detail/amd_gfx1250_TDM.h>)
#define GFX1250_TDM_6ARG 1
#endif
#endif
#ifndef GFX1250_TDM_6ARG
#define GFX1250_TDM_6ARG 0
#endif

__device__ __forceinline__ void tdm_load_2d(
    u32 lds_byte, const void* gptr,
    u32 tile_d0, u32 tile_d1, u32 stride0_elems,
    u32 pad_interval_code, u32 pad_amount_code)
{
    const unsigned long long ga = (unsigned long long)(uintptr_t)gptr;
    u32x4 g0;
    g0[0] = 1u;                                                // count=1 (valid)
    g0[1] = lds_byte;                                          // lds_addr
    g0[2] = (u32)ga;                                           // global_addr lo
    g0[3] = (u32)((ga >> 32) & 0x01ffffffu) | (2u << 30);      // addr hi | type=2
    i32x8 g1;
    g1[0] = (int)((1u << 16) |                                 // data_size = 2B
                  (1u << 20) |                                 // pad_enable
                  (pad_interval_code << 22) | (pad_amount_code << 25));
    g1[1] = (int)((tile_d0 & 0xffffu) << 16);                  // tensor_dim0 lo16 @48
    g1[2] = (int)(((tile_d0 >> 16) & 0xffffu) | ((tile_d1 & 0xffffu) << 16));
    g1[3] = (int)(((tile_d1 >> 16) & 0xffffu) | ((tile_d0 & 0xffffu) << 16)); // tile_dim0
    g1[4] = (int)(tile_d1 & 0xffffu);                          // tile_dim1, tile_dim2=0
    g1[5] = (int)stride0_elems;                                // tensor_dim0_stride lo32
    g1[6] = 0;
    g1[7] = 0;
    i32x4 z4 = {0, 0, 0, 0};
#if GFX1250_TDM_6ARG
    i32x8 z8 = {0, 0, 0, 0, 0, 0, 0, 0};
    __builtin_amdgcn_tensor_load_to_lds(g0, g1, z4, z4, z8, 0);
#else
    __builtin_amdgcn_tensor_load_to_lds(g0, g1, z4, z4, 0);
#endif
}

// ---------------------------------------------------------------------------
// Double-buffered GEMM: C[M,N] = A[M,K] @ B[K,N] + bias. f16 WMMA, f32 accum.
// Block tile 128x128, BK=32, 256 threads = 8 waves (4m x 2n), wave tile 32x64.
// f32 A: reg-staged loads overlap WMMA; f16 A: staged by the TDM while the
// waves compute. B tile loaded down columns and stored transposed ([n][k]).
// NOTE: LDS buffer bases are computed arithmetically from `buf` — storing LDS
// pointers in an indexed const array creates an addrspacecast static
// initializer that ld.lld rejects.
// ---------------------------------------------------------------------------
template <typename AT, bool OUT_F16>
__global__ __launch_bounds__(256) void gemm_wmma(
    const AT* __restrict__ A, const float* __restrict__ Bm,
    const float* __restrict__ bias, void* __restrict__ Cout,
    int M, int N, int K)
{
    constexpr int TILE_H = 128 * 40;              // halves per buffer
    __shared__ _Float16 smem[4 * TILE_H];         // As0 As1 Bs0 Bs1

    const int tid  = threadIdx.x;
    const int lane = tid & 31;
    const int wave = tid >> 5;
    const int wm   = wave >> 1;                   // 0..3
    const int wn   = wave & 1;                    // 0..1
    const int half = lane >> 4;
    const int l16  = lane & 15;
    const int row0 = blockIdx.y * 128;
    const int col0 = blockIdx.x * 128;

    v8f c[2][4] = {};

    auto Asp = [&](int buf) -> _Float16* { return smem + buf * TILE_H; };
    auto Bsp = [&](int buf) -> _Float16* { return smem + (2 + buf) * TILE_H; };

    auto load_a = [&](int kk, v4f (&ra)[4]) {
        #pragma unroll
        for (int p = 0; p < 4; p++) {
            const int idx = tid + p * 256;        // 0..1023
            const int r = idx >> 3, c4 = idx & 7;
            ra[p] = *(const v4f*)((const float*)A + (size_t)(row0 + r) * K + kk + c4 * 4);
        }
    };
    auto store_a = [&](int buf, const v4f (&ra)[4]) {
        _Float16* As = Asp(buf);
        #pragma unroll
        for (int p = 0; p < 4; p++) {
            const int idx = tid + p * 256;
            const int r = idx >> 3, c4 = idx & 7;
            v4h h;
            #pragma unroll
            for (int i = 0; i < 4; i++) h[i] = (_Float16)ra[p][i];
            *(v4h*)&As[r * 40 + c4 * 4] = h;
        }
    };
    auto load_b = [&](int kk, v4f (&rb)[4]) {
        #pragma unroll
        for (int p = 0; p < 4; p++) {
            const int idx = tid + p * 256;        // 0..1023
            const int n = idx & 127, kg = idx >> 7;   // kg 0..7
            v4f t;
            #pragma unroll
            for (int j = 0; j < 4; j++)
                t[j] = Bm[(size_t)(kk + kg * 4 + j) * N + col0 + n];
            rb[p] = t;
        }
    };
    auto store_b = [&](int buf, const v4f (&rb)[4]) {
        _Float16* Bs = Bsp(buf);
        #pragma unroll
        for (int p = 0; p < 4; p++) {
            const int idx = tid + p * 256;
            const int n = idx & 127, kg = idx >> 7;
            v4h h;
            #pragma unroll
            for (int i = 0; i < 4; i++) h[i] = (_Float16)rb[p][i];
            *(v4h*)&Bs[n * 40 + kg * 4] = h;
        }
    };
    auto compute = [&](int buf) {
        _Float16* As = Asp(buf);
        _Float16* Bs = Bsp(buf);
        v16h a[2];
        #pragma unroll
        for (int mi = 0; mi < 2; mi++) {
            const _Float16* pa = &As[(wm * 32 + mi * 16 + l16) * 40];
            const v8h lo = *(const v8h*)(pa + 16 * half);
            const v8h hi = *(const v8h*)(pa + 16 * half + 8);
            #pragma unroll
            for (int e = 0; e < 8; e++) { a[mi][e] = lo[e]; a[mi][8 + e] = hi[e]; }
        }
        #pragma unroll
        for (int ni = 0; ni < 4; ni++) {
            const _Float16* pb = &Bs[(wn * 64 + ni * 16 + l16) * 40 + 16 * half];
            const v8h lo = *(const v8h*)pb;
            const v8h hi = *(const v8h*)(pb + 8);
            v16h b;
            #pragma unroll
            for (int e = 0; e < 8; e++) { b[e] = lo[e]; b[8 + e] = hi[e]; }
            #pragma unroll
            for (int mi = 0; mi < 2; mi++)
                c[mi][ni] = __builtin_amdgcn_wmma_f32_16x16x32_f16(
                    false, a[mi], false, b, (short)0, c[mi][ni], false, false);
        }
    };

    // ---- prologue: fill buffer 0 ----
    {
        v4f ra[4], rb[4];
        if constexpr (sizeof(AT) == 4) load_a(0, ra);
        else if (wave == 0) tdm_load_2d(0u, (const _Float16*)A + (size_t)row0 * K,
                                        32, 128, (u32)K, 3, 3);
        load_b(0, rb);
        if constexpr (sizeof(AT) == 4) store_a(0, ra);
        store_b(0, rb);
        if constexpr (sizeof(AT) == 2) {
            if (wave == 0) __builtin_amdgcn_s_wait_tensorcnt(0);
        }
    }
    __syncthreads();

    int buf = 0;
    for (int k0 = 0; k0 < K; k0 += 32) {
        v4f ra[4], rb[4];
        const bool hn = (k0 + 32 < K);
        if (hn) {
            if constexpr (sizeof(AT) == 4) load_a(k0 + 32, ra);
            else if (wave == 0)
                tdm_load_2d((u32)((buf ^ 1) * TILE_H * 2),
                            (const _Float16*)A + (size_t)row0 * K + k0 + 32,
                            32, 128, (u32)K, 3, 3);
            load_b(k0 + 32, rb);
            if (k0 + 64 < K) {                    // warm L2 for the tile after next
                __builtin_prefetch((const char*)(A + (size_t)(row0 + (tid & 127)) * K + k0 + 64), 0, 0);
                __builtin_prefetch((const char*)(Bm + (size_t)(k0 + 64 + (tid >> 3)) * N + col0 + (tid & 7) * 16), 0, 0);
            }
        }
        compute(buf);
        if (hn) {
            if constexpr (sizeof(AT) == 4) store_a(buf ^ 1, ra);
            store_b(buf ^ 1, rb);
            if constexpr (sizeof(AT) == 2) {
                if (wave == 0) __builtin_amdgcn_s_wait_tensorcnt(0);
            }
        }
        __syncthreads();
        buf ^= 1;
    }

    // ---- epilogue ----
    #pragma unroll
    for (int mi = 0; mi < 2; mi++) {
        #pragma unroll
        for (int ni = 0; ni < 4; ni++) {
            const int ccol = col0 + wn * 64 + ni * 16 + l16;
            const float bb = bias[ccol];
            const int rbase = row0 + wm * 32 + mi * 16 + 8 * half;
            #pragma unroll
            for (int v = 0; v < 8; v++) {
                const float val = c[mi][ni][v] + bb;
                const size_t off = (size_t)(rbase + v) * N + ccol;
                if constexpr (OUT_F16) ((_Float16*)Cout)[off] = (_Float16)val;
                else                   ((float*)Cout)[off]    = val;
            }
        }
    }
}

// ---------------------------------------------------------------------------
// RoPE + group-sum of q heads; head-major k (rope'd); v stored TRANSPOSED
// per head: vt[b][h][d][t] so attention V staging is a plain 2D tile copy.
// ---------------------------------------------------------------------------
__global__ __launch_bounds__(256) void rope_pack(
    const _Float16* __restrict__ qh, const _Float16* __restrict__ kh,
    const _Float16* __restrict__ vh,
    _Float16* __restrict__ qs, _Float16* __restrict__ kr, _Float16* __restrict__ vt)
{
    constexpr int T = 2048, NKV = 4, D = 128, C = 2048, CKV = 512;
    const int idx = blockIdx.x * 256 + threadIdx.x;
    const int i  = idx & 63;
    const int t  = (idx >> 6) & (T - 1);
    const int bh = idx >> 17;
    const int h  = bh & 3;
    const int b  = bh >> 2;

    const float ang = (float)t * __expf(-(float)i * 0.14391156831212787f); // 10000^(-i/64)
    float sn, cs;
    sincosf(ang, &sn, &cs);

    const size_t rowq = (size_t)(b * T + t) * C;
    float s1 = 0.f, s2 = 0.f;
    #pragma unroll
    for (int gi = 0; gi < 4; gi++) {
        const int hq = h * 4 + gi;
        s1 += (float)qh[rowq + hq * D + 2 * i];
        s2 += (float)qh[rowq + hq * D + 2 * i + 1];
    }
    const size_t ob = ((size_t)(b * NKV + h) * T + t) * D;
    qs[ob + i]      = (_Float16)(s1 * cs - s2 * sn);
    qs[ob + 64 + i] = (_Float16)(s1 * sn + s2 * cs);

    const size_t rowk = (size_t)(b * T + t) * CKV + h * D;
    const float k1 = (float)kh[rowk + 2 * i];
    const float k2 = (float)kh[rowk + 2 * i + 1];
    kr[ob + i]      = (_Float16)(k1 * cs - k2 * sn);
    kr[ob + 64 + i] = (_Float16)(k1 * sn + k2 * cs);

    const size_t vhead = (size_t)(b * NKV + h) * D * T;
    vt[vhead + (size_t)(2 * i) * T + t]     = vh[rowk + 2 * i];
    vt[vhead + (size_t)(2 * i + 1) * T + t] = vh[rowk + 2 * i + 1];
}

// ---------------------------------------------------------------------------
// Flash attention (causal), TDM-staged K/V with software pipelining.
// One block per (b, kv-head, 64 query rows); 4 waves x 16 rows.
// Q fragments live in registers (read once from global).
// Pipeline: K_{j+1} DMA overlaps O-GEMM_j; V_{j+1} overlaps S_{j+1}+softmax.
// ---------------------------------------------------------------------------
__global__ __launch_bounds__(128) void flash_attn(
    const _Float16* __restrict__ qs, const _Float16* __restrict__ kr,
    const _Float16* __restrict__ vt, _Float16* __restrict__ yh)
{
    constexpr int T = 2048, D = 128, NKV = 4, CKV = 512;
    __shared__ _Float16 smem[64 * 136 + 128 * 72 + 4 * 16 * 72];
    _Float16* const Ks  = smem;                       // byte 0,     [key][d] stride 136
    _Float16* const Vst = smem + 64 * 136;            // byte 17408, [d][key] stride 72
    _Float16* const Ps  = smem + 64 * 136 + 128 * 72; // per-wave P [m][key] stride 72
    constexpr u32 KS_B = 0, VS_B = 64 * 136 * 2;

    const int tid  = threadIdx.x;
    const int lane = tid & 31;
    const int wave = tid >> 5;
    const int half = lane >> 4;
    const int l16  = lane & 15;
    const int qtile = blockIdx.x;
    const int h     = blockIdx.y;
    const int b     = blockIdx.z;
    const size_t headoff = (size_t)(b * NKV + h) * T * D;   // q/k head base
    const size_t vhead   = (size_t)(b * NKV + h) * D * T;   // v head base ([d][t])

    // ---- Q fragments in registers (A layout) ----
    v16h qa[4];
    const _Float16* qbase = qs + headoff + (size_t)(qtile * 64 + wave * 16 + l16) * D;
    #pragma unroll
    for (int kk = 0; kk < 4; kk++) {
        const v8h lo = *(const v8h*)(qbase + kk * 32 + 16 * half);
        const v8h hi = *(const v8h*)(qbase + kk * 32 + 16 * half + 8);
        #pragma unroll
        for (int e = 0; e < 8; e++) { qa[kk][e] = lo[e]; qa[kk][8 + e] = hi[e]; }
    }

    v8f o[8] = {};
    float mrow[8], lrow[8];
    const float NEGINF = -__builtin_inff();
    #pragma unroll
    for (int v = 0; v < 8; v++) { mrow[v] = NEGINF; lrow[v] = 0.f; }
    const float sc = 0.08838834764831845f;            // 1/sqrt(128)

    // K tile: 64 rows x 128 elems, row stride 128; pad 4 DW after each 64 DW row
    // V tile: 128 rows x 64 elems, row stride T;   pad 4 DW after each 32 DW row
    if (wave == 0) {
        tdm_load_2d(KS_B, kr + headoff, 128, 64, (u32)D, 5, 3);
        tdm_load_2d(VS_B, vt + vhead,    64, 128, (u32)T, 4, 3);
    }

    for (int jt = 0; jt <= qtile; jt++) {
        const int kb0 = jt * 64;
        if (wave == 0) __builtin_amdgcn_s_wait_tensorcnt(1);  // K_jt arrived (in-order)
        __syncthreads();

        // ---- S = Q K^T (16x64 per wave) ----
        v8f s[4] = {};
        #pragma unroll
        for (int kk = 0; kk < 4; kk++) {
            #pragma unroll
            for (int ni = 0; ni < 4; ni++) {
                const _Float16* pb = &Ks[(ni * 16 + l16) * 136 + kk * 32 + 16 * half];
                const v8h lo = *(const v8h*)pb;
                const v8h hi = *(const v8h*)(pb + 8);
                v16h bf;
                #pragma unroll
                for (int e = 0; e < 8; e++) { bf[e] = lo[e]; bf[8 + e] = hi[e]; }
                s[ni] = __builtin_amdgcn_wmma_f32_16x16x32_f16(
                    false, qa[kk], false, bf, (short)0, s[ni], false, false);
            }
        }

        // ---- scale + causal mask + online softmax ----
        const int rowg0 = qtile * 64 + wave * 16 + 8 * half;
        _Float16* pw = &Ps[wave * 16 * 72];
        #pragma unroll
        for (int v = 0; v < 8; v++) {
            float mx = NEGINF;
            #pragma unroll
            for (int ni = 0; ni < 4; ni++) {
                const int kg = kb0 + ni * 16 + l16;
                float val = s[ni][v] * sc;
                val = (kg <= rowg0 + v) ? val : NEGINF;
                s[ni][v] = val;
                mx = fmaxf(mx, val);
            }
            #pragma unroll
            for (int off = 1; off < 16; off <<= 1)
                mx = fmaxf(mx, __shfl_xor(mx, off, 32));
            const float mnew = fmaxf(mrow[v], mx);
            const float alpha = __expf(mrow[v] - mnew);
            float ps = 0.f;
            #pragma unroll
            for (int ni = 0; ni < 4; ni++) {
                const float e = __expf(s[ni][v] - mnew);
                s[ni][v] = e;
                ps += e;
            }
            #pragma unroll
            for (int off = 1; off < 16; off <<= 1)
                ps += __shfl_xor(ps, off, 32);
            lrow[v] = lrow[v] * alpha + ps;
            mrow[v] = mnew;
            #pragma unroll
            for (int f = 0; f < 8; f++) o[f][v] *= alpha;
            #pragma unroll
            for (int ni = 0; ni < 4; ni++)
                pw[(v + 8 * half) * 72 + ni * 16 + l16] = (_Float16)s[ni][v];
        }

        if (wave == 0) __builtin_amdgcn_s_wait_tensorcnt(0);  // V_jt arrived
        __syncthreads();   // V visible to all; all waves done reading K_jt
        if (wave == 0 && jt + 1 <= qtile)                      // K_{jt+1} overlaps O-GEMM
            tdm_load_2d(KS_B, kr + headoff + (size_t)(jt + 1) * 64 * D, 128, 64, (u32)D, 5, 3);

        // ---- O += P V ----
        #pragma unroll
        for (int kk2 = 0; kk2 < 2; kk2++) {
            v16h a2;
            const _Float16* pa = &pw[l16 * 72 + kk2 * 32 + 16 * half];
            const v8h lo = *(const v8h*)pa;
            const v8h hi = *(const v8h*)(pa + 8);
            #pragma unroll
            for (int e = 0; e < 8; e++) { a2[e] = lo[e]; a2[8 + e] = hi[e]; }
            #pragma unroll
            for (int f = 0; f < 8; f++) {
                const _Float16* pb = &Vst[(f * 16 + l16) * 72 + kk2 * 32 + 16 * half];
                const v8h blo = *(const v8h*)pb;
                const v8h bhi = *(const v8h*)(pb + 8);
                v16h bf;
                #pragma unroll
                for (int e = 0; e < 8; e++) { bf[e] = blo[e]; bf[8 + e] = bhi[e]; }
                o[f] = __builtin_amdgcn_wmma_f32_16x16x32_f16(
                    false, a2, false, bf, (short)0, o[f], false, false);
            }
        }
        __syncthreads();   // all done reading V_jt
        if (wave == 0 && jt + 1 <= qtile)                      // V_{jt+1} overlaps next S
            tdm_load_2d(VS_B, vt + vhead + (size_t)(jt + 1) * 64, 64, 128, (u32)T, 4, 3);
    }

    // ---- normalize and store y[b][t][h*128 + dcol] ----
    const int trow0 = qtile * 64 + wave * 16 + 8 * half;
    #pragma unroll
    for (int f = 0; f < 8; f++) {
        #pragma unroll
        for (int v = 0; v < 8; v++) {
            const float val = o[f][v] / lrow[v];
            const int t = trow0 + v;
            yh[(size_t)(b * T + t) * CKV + h * 128 + f * 16 + l16] = (_Float16)val;
        }
    }
}

// ---------------------------------------------------------------------------
extern "C" void kernel_launch(void* const* d_in, const int* in_sizes, int n_in,
                              void* d_out, int out_size, void* d_ws, size_t ws_size,
                              hipStream_t stream)
{
    const float* x  = (const float*)d_in[0];
    const float* Wq = (const float*)d_in[1];
    const float* bq = (const float*)d_in[2];
    const float* Wk = (const float*)d_in[3];
    const float* bk = (const float*)d_in[4];
    const float* Wv = (const float*)d_in[5];
    const float* bv = (const float*)d_in[6];
    const float* Wo = (const float*)d_in[7];
    const float* bo = (const float*)d_in[8];
    float* out = (float*)d_out;

    constexpr int B = 4, T = 2048, C = 2048, CKV = 512, NKV = 4;
    const int M = B * T;                          // 8192

    _Float16* qh = (_Float16*)d_ws;               // M*C
    _Float16* kh = qh + (size_t)M * C;            // M*CKV
    _Float16* vh = kh + (size_t)M * CKV;          // M*CKV
    _Float16* qsum  = vh + (size_t)M * CKV;       // M*CKV  (b,h,t,d)
    _Float16* krope = qsum + (size_t)M * CKV;     // M*CKV  (b,h,t,d)
    _Float16* vtr   = krope + (size_t)M * CKV;    // M*CKV  (b,h,d,t)
    _Float16* yh = vtr + (size_t)M * CKV;         // M*CKV

    gemm_wmma<float, true><<<dim3(C / 128, M / 128), 256, 0, stream>>>(x, Wq, bq, qh, M, C, C);
    gemm_wmma<float, true><<<dim3(CKV / 128, M / 128), 256, 0, stream>>>(x, Wk, bk, kh, M, CKV, C);
    gemm_wmma<float, true><<<dim3(CKV / 128, M / 128), 256, 0, stream>>>(x, Wv, bv, vh, M, CKV, C);

    rope_pack<<<(B * NKV * T * 64) / 256, 256, 0, stream>>>(qh, kh, vh, qsum, krope, vtr);

    flash_attn<<<dim3(T / 64, NKV, B), 128, 0, stream>>>(qsum, krope, vtr, yh);

    gemm_wmma<_Float16, false><<<dim3(C / 128, M / 128), 256, 0, stream>>>(yh, Wo, bo, out, M, C, CKV);
}
